// ConditionalFeedForward_70901320122871
// MI455X (gfx1250) — compile-verified, matrix operand order checked by reading, not run on hardware
//
#include <hip/hip_runtime.h>
#include <hip/hip_bf16.h>

// Problem constants (match reference)
#define TOKENS 2048
#define DMODEL 1024
#define IDIM   4096
#define NEXP   8
#define TOPK   2
#define CAP    (TOKENS * TOPK)   // worst-case tokens routed to one expert
#define MTILE  32                // tokens per block (2 x 16-row WMMA subtiles)
#define ICHUNK 512               // I processed per chunk (h chunk kept in LDS)

// Dynamic LDS: sX (MTILE x DMODEL bf16) | sH (MTILE x ICHUNK bf16) | tok | wt
#define SMEM_BYTES (MTILE * DMODEL * 2 + MTILE * ICHUNK * 2 + MTILE * 8)

typedef __attribute__((ext_vector_type(16))) __bf16 v16bf;
typedef __attribute__((ext_vector_type(8)))  __bf16 v8bf;
typedef __attribute__((ext_vector_type(8)))  float  v8f;

__device__ __forceinline__ v8f wmma_bf16(v16bf a, v16bf b, v8f c) {
    return __builtin_amdgcn_wmma_f32_16x16x32_bf16(
        /*neg_a=*/false, a, /*neg_b=*/false, b,
        /*c_mod=*/(short)0, c, /*reuse_a=*/false, /*reuse_b=*/false);
}

// A fragment (16x32 bf16) from an LDS tile [16 rows x row_stride bf16].
// Lane M = lane&15, half h = lane>>4.
// Elements 0..7  hold K = h*8 + 0..7; elements 8..15 hold K = 16 + h*8 + 0..7
// => two 16-byte ds_load_b128 per lane.
__device__ __forceinline__ v16bf load_a_lds(const __bf16* s, int row_stride,
                                            int kbase, int lane) {
    const int M = lane & 15;
    const int h = lane >> 4;
    const __bf16* p = s + M * row_stride + kbase + (h << 3);
    const v8bf lo = *(const v8bf*)(p);
    const v8bf hi = *(const v8bf*)(p + 16);
    v16bf a;
#pragma unroll
    for (int e = 0; e < 8; ++e) { a[e] = lo[e]; a[e + 8] = hi[e]; }
    return a;
}

// B fragment (32x16 bf16) from fp32 weight rows in global memory.
// B[k][n] = W[nbase+n][kbase+k]; lane N = lane&15, half h = lane>>4.
// Four global_load_b128 + packed v_cvt_pk_bf16_f32 conversion.
__device__ __forceinline__ v16bf load_b_global(const float* __restrict__ base,
                                               int row_stride, int nbase,
                                               int kbase, int lane) {
    const int N = lane & 15;
    const int h = lane >> 4;
    const float* p = base + (long)(nbase + N) * row_stride + kbase + (h << 3);
    const float4 f0 = *(const float4*)(p);
    const float4 f1 = *(const float4*)(p + 4);
    const float4 f2 = *(const float4*)(p + 16);
    const float4 f3 = *(const float4*)(p + 20);
    v16bf b;
    b[0]  = (__bf16)f0.x; b[1]  = (__bf16)f0.y;
    b[2]  = (__bf16)f0.z; b[3]  = (__bf16)f0.w;
    b[4]  = (__bf16)f1.x; b[5]  = (__bf16)f1.y;
    b[6]  = (__bf16)f1.z; b[7]  = (__bf16)f1.w;
    b[8]  = (__bf16)f2.x; b[9]  = (__bf16)f2.y;
    b[10] = (__bf16)f2.z; b[11] = (__bf16)f2.w;
    b[12] = (__bf16)f3.x; b[13] = (__bf16)f3.y;
    b[14] = (__bf16)f3.z; b[15] = (__bf16)f3.w;
    return b;
}

// ---------------------------------------------------------------------------
// Kernel 0: zero output + per-expert counters (must run every call)
// ---------------------------------------------------------------------------
__global__ void moe_zero_kernel(float* __restrict__ out, int n,
                                int* __restrict__ counts) {
    const int i = blockIdx.x * blockDim.x + threadIdx.x;
    for (int j = i; j < n; j += gridDim.x * blockDim.x) out[j] = 0.0f;
    if (i < NEXP) counts[i] = 0;
}

// ---------------------------------------------------------------------------
// Kernel 1: route (token, gate weight) pairs into per-expert compact lists.
// Output remains deterministic: each out[t,d] gets exactly its token's K=2
// contributions and 2-operand fp add is commutative.
// ---------------------------------------------------------------------------
__global__ void moe_route_kernel(const int* __restrict__ idx,
                                 const float* __restrict__ gatew,
                                 int* __restrict__ counts,
                                 int* __restrict__ tokList,
                                 float* __restrict__ wtList) {
    const int i = blockIdx.x * blockDim.x + threadIdx.x;
    if (i >= TOKENS * TOPK) return;
    const int e   = idx[i];
    const int tok = i / TOPK;
    const int pos = atomicAdd(&counts[e], 1);
    tokList[e * CAP + pos] = tok;
    wtList[e * CAP + pos]  = gatew[i];
}

// ---------------------------------------------------------------------------
// Kernel 2: per-(expert, 32-token tile) fused SwiGLU FFN via bf16 WMMA.
// Every B fragment feeds 2 M-subtiles; every A fragment feeds >=2 WMMAs.
// ---------------------------------------------------------------------------
__global__ __launch_bounds__(256)
void moe_gemm_kernel(const float* __restrict__ x,
                     const float* __restrict__ w1,
                     const float* __restrict__ w2,
                     const float* __restrict__ w3,
                     const int* __restrict__ counts,
                     const int* __restrict__ tokList,
                     const float* __restrict__ wtList,
                     float* __restrict__ out) {
    extern __shared__ __align__(16) char smem[];
    __bf16* sX   = (__bf16*)smem;                    // MTILE x DMODEL (64 KB)
    __bf16* sH   = sX + MTILE * DMODEL;              // MTILE x ICHUNK (32 KB)
    int*    sTok = (int*)(sH + MTILE * ICHUNK);
    float*  sWt  = (float*)(sTok + MTILE);

    const int e     = blockIdx.y;        // expert slow-varying -> L2 locality
    const int tbase = blockIdx.x * MTILE;
    const int cnt   = counts[e];
    if (tbase >= cnt) return;            // uniform early-exit

    const float* w1e = w1 + (long)e * IDIM * DMODEL;
    const float* w3e = w3 + (long)e * IDIM * DMODEL;
    const float* w2e = w2 + (long)e * DMODEL * IDIM;

    if (threadIdx.x < MTILE) {
        const int p = tbase + threadIdx.x;
        if (p < cnt) {
            sTok[threadIdx.x] = tokList[e * CAP + p];
            sWt[threadIdx.x]  = wtList[e * CAP + p];
        } else {
            sTok[threadIdx.x] = -1;
            sWt[threadIdx.x]  = 0.0f;
        }
    }
    __syncthreads();

    // Stage x tile -> LDS as bf16 (8 threads per row, 128 floats each)
    {
        const int row  = threadIdx.x >> 3;           // 0..31
        const int col0 = (threadIdx.x & 7) * 128;
        const int t    = sTok[row];
        const float* xr = x + (long)(t < 0 ? 0 : t) * DMODEL;
#pragma unroll
        for (int i = 0; i < 128; i += 4) {
            float4 v = (t >= 0) ? *(const float4*)(xr + col0 + i)
                                : make_float4(0.f, 0.f, 0.f, 0.f);
            __bf16* d = &sX[row * DMODEL + col0 + i];
            d[0] = (__bf16)v.x; d[1] = (__bf16)v.y;
            d[2] = (__bf16)v.z; d[3] = (__bf16)v.w;
        }
    }
    __syncthreads();

    const int wave = threadIdx.x >> 5;
    const int lane = threadIdx.x & 31;
    const int N    = lane & 15;
    const int half = lane >> 4;

    v8f yacc[8][2] = {};   // 8 D-tiles x 2 M-subtiles, persistent f32 C tiles

    for (int c = 0; c < IDIM / ICHUNK; ++c) {
        const int ibase0 = c * ICHUNK;

        // ---- Phase 1: h = silu(x*w1^T) * (x*w3^T) for this I-chunk ----
#pragma unroll 1
        for (int j = 0; j < 4; ++j) {
            const int ntl = wave * 4 + j;            // chunk-local N tile 0..31
            const int ng  = ibase0 + ntl * 16;       // global I base of tile
            v8f a1m0 = {}, a1m1 = {}, a3m0 = {}, a3m1 = {};
#pragma unroll 1
            for (int kb = 0; kb < DMODEL; kb += 32) {
                const v16bf a0 = load_a_lds(sX,               DMODEL, kb, lane);
                const v16bf a1 = load_a_lds(sX + 16 * DMODEL, DMODEL, kb, lane);
                const v16bf b1 = load_b_global(w1e, DMODEL, ng, kb, lane);
                const v16bf b3 = load_b_global(w3e, DMODEL, ng, kb, lane);
                a1m0 = wmma_bf16(a0, b1, a1m0);
                a1m1 = wmma_bf16(a1, b1, a1m1);
                a3m0 = wmma_bf16(a0, b3, a3m0);
                a3m1 = wmma_bf16(a1, b3, a3m1);
            }
#pragma unroll
            for (int m = 0; m < 2; ++m) {
                const v8f g1 = m ? a1m1 : a1m0;
                const v8f g3 = m ? a3m1 : a3m0;
#pragma unroll
                for (int r = 0; r < 8; ++r) {
                    const int row = m * 16 + r + (half << 3);
                    const float g = g1[r];
                    // silu via hardware reciprocal (v_rcp_f32)
                    const float s = g * __builtin_amdgcn_rcpf(1.0f + __expf(-g));
                    sH[row * ICHUNK + ntl * 16 + N] = (__bf16)(s * g3[r]);
                }
            }
        }
        __syncthreads();

        // ---- Phase 2: y += h_chunk * w2^T (each wave owns 8 D tiles) ----
#pragma unroll 1
        for (int kk = 0; kk < ICHUNK; kk += 32) {
            const v16bf a0 = load_a_lds(sH,               ICHUNK, kk, lane);
            const v16bf a1 = load_a_lds(sH + 16 * ICHUNK, ICHUNK, kk, lane);
#pragma unroll
            for (int j = 0; j < 8; ++j) {
                const int dt = wave * 8 + j;         // D tile 0..63
                const v16bf b = load_b_global(w2e, IDIM, dt * 16,
                                              ibase0 + kk, lane);
                yacc[j][0] = wmma_bf16(a0, b, yacc[j][0]);
                yacc[j][1] = wmma_bf16(a1, b, yacc[j][1]);
            }
        }
        __syncthreads();
    }

    // ---- Scatter: out[tok, d] += gate * y ----
#pragma unroll
    for (int j = 0; j < 8; ++j) {
        const int d0 = (wave * 8 + j) * 16 + N;
#pragma unroll
        for (int m = 0; m < 2; ++m) {
#pragma unroll
            for (int r = 0; r < 8; ++r) {
                const int row = m * 16 + r + (half << 3);
                const int t = sTok[row];
                if (t >= 0) {
                    const float val = yacc[j][m][r] * sWt[row];
                    unsafeAtomicAdd(out + (long)t * DMODEL + d0, val);
                }
            }
        }
    }
}

// ---------------------------------------------------------------------------
// Host-side launcher
// ---------------------------------------------------------------------------
extern "C" void kernel_launch(void* const* d_in, const int* in_sizes, int n_in,
                              void* d_out, int out_size, void* d_ws, size_t ws_size,
                              hipStream_t stream) {
    const float* x    = (const float*)d_in[0];  // [T, D]
    const int*   eidx = (const int*)  d_in[1];  // [T, K]
    const float* ew   = (const float*)d_in[2];  // [T, K]
    const float* w1   = (const float*)d_in[3];  // [E, I, D]
    const float* w2   = (const float*)d_in[4];  // [E, D, I]
    const float* w3   = (const float*)d_in[5];  // [E, I, D]
    float* out = (float*)d_out;                 // [T, D]

    // Workspace layout: counts[64-pad] | tokList[E*CAP] | wtList[E*CAP]
    int*   counts  = (int*)d_ws;
    int*   tokList = counts + 64;
    float* wtList  = (float*)(tokList + NEXP * CAP);

    moe_zero_kernel<<<1024, 256, 0, stream>>>(out, TOKENS * DMODEL, counts);

    moe_route_kernel<<<(TOKENS * TOPK + 255) / 256, 256, 0, stream>>>(
        eidx, ew, counts, tokList, wtList);

    dim3 grid(CAP / MTILE, NEXP);  // tile fast, expert slow -> weights stay in L2
    moe_gemm_kernel<<<grid, 256, SMEM_BYTES, stream>>>(
        x, w1, w2, w3, counts, tokList, wtList, out);
}
